// CoordGNN_29180007809228
// MI455X (gfx1250) — matrix operand order repaired and checked by reading, compiled, not run on hardware
//
#include <hip/hip_runtime.h>

// ---------------------------------------------------------------------------
// CoordGNN on gfx1250: fused edge-MLP (TDM-staged weights) + WMMA GEMM.
// wave32, f16 WMMA with f32 accumulate, pre-transposed f16 weights.
// ---------------------------------------------------------------------------

typedef __attribute__((ext_vector_type(16))) _Float16 v16h;
typedef __attribute__((ext_vector_type(8)))  float    v8f;
typedef unsigned int uint32x4 __attribute__((ext_vector_type(4)));
typedef unsigned int uint32x8 __attribute__((ext_vector_type(8)));

#define N0_  100000
#define N1_  50000
#define N2_  25000
#define E0_  1600000
#define E1_  800000
#define INF_ 16
#define H_   128

__device__ __forceinline__ float elu_(float x) {
    return x > 0.f ? x : (__expf(x) - 1.f);
}

union HalfPack { unsigned u; _Float16 h[2]; };
union H1 { unsigned short s; _Float16 h; };

// ---------------------------------------------------------------------------
// TDM: stage a 2D f16 matrix (rows x 128 halves, row-major, contiguous) into
// LDS with a 1-dword pad after every 64 dwords -> LDS row stride 65 dwords.
// D# layout per CDNA5 ISA 8.3/8.4.
// ---------------------------------------------------------------------------
__device__ __forceinline__ void tdm_load_w2_f16(unsigned ldsOff, const void* gptr,
                                                unsigned rows) {
    unsigned long long ga = (unsigned long long)gptr;
    const unsigned d0 = 128u;          // halves per row (tensor/tile dim0)
    const unsigned d1 = rows;          // rows (tensor/tile dim1)
    uint32x4 g0;
    g0.x = 1u;                                         // count=1 (valid user D#)
    g0.y = ldsOff;                                     // lds_addr
    g0.z = (unsigned)(ga & 0xFFFFFFFFu);               // global_addr[31:0]
    g0.w = (unsigned)((ga >> 32) & 0x01FFFFFFu)        // global_addr[56:32]
         | (2u << 30);                                 // type = 2 (image)
    uint32x8 g1;
    g1.s0 = (1u << 16)                                 // data_size = 2 bytes
          | (1u << 20)                                 // pad_enable
          | (5u << 22);                                // pad_interval = 64 dwords
                                                       // pad_amount = 1 dword (code 0)
    g1.s1 = (d0 & 0xFFFFu) << 16;                      // tensor_dim0 [63:48]
    g1.s2 = (d0 >> 16) | ((d1 & 0xFFFFu) << 16);       // dim0 hi | tensor_dim1 lo
    g1.s3 = (d1 >> 16) | (d0 << 16);                   // dim1 hi | tile_dim0
    g1.s4 = d1;                                        // tile_dim1 (tile_dim2 = 0)
    g1.s5 = d0;                                        // tensor_dim0_stride lo
    g1.s6 = 0u;
    g1.s7 = 0u;
    asm volatile("tensor_load_to_lds %0, %1" :: "s"(g0), "s"(g1) : "memory");
}

// ---------------------------------------------------------------------------
// Weight pre-pass: W (K x N f32, row-major) -> Wt (N x Kpad f16, zero padded)
// ---------------------------------------------------------------------------
__global__ void wconv_kernel(const float* __restrict__ W,
                             unsigned short* __restrict__ Wt,
                             int K, int N, int Kpad) {
    int t = blockIdx.x * 256 + threadIdx.x;
    if (t >= N * Kpad) return;
    int n = t / Kpad, k = t - n * Kpad;
    float v = (k < K) ? W[k * N + n] : 0.f;
    H1 c; c.h = (_Float16)v;
    Wt[t] = c.s;
}

// ---------------------------------------------------------------------------
// Segment softmax over edge scores
// ---------------------------------------------------------------------------
__global__ void score_max_kernel(const float2* __restrict__ off,
                                 const int* __restrict__ dst,
                                 unsigned* __restrict__ smax, int E) {
    int e = blockIdx.x * 256 + threadIdx.x;
    if (e >= E) return;
    float2 o = off[e];
    float score = 1.f / (fabsf(o.x) + fabsf(o.y) + 0.001f);
    atomicMax(&smax[dst[e]], __float_as_uint(score));  // positive: uint order == float order
}

__global__ void score_exp_kernel(const float2* __restrict__ off,
                                 const int* __restrict__ dst,
                                 const unsigned* __restrict__ smax,
                                 float* __restrict__ denom,
                                 float* __restrict__ ex, int E) {
    int e = blockIdx.x * 256 + threadIdx.x;
    if (e >= E) return;
    float2 o = off[e];
    float score = 1.f / (fabsf(o.x) + fabsf(o.y) + 0.001f);
    float x = __expf(score - __uint_as_float(smax[dst[e]]));
    ex[e] = x;
    atomicAdd(&denom[dst[e]], x);
}

// ---------------------------------------------------------------------------
// Fused edge conv level 0 (in_f = 16): one wave = 16 edges, 3 convs fused.
// Layer1 (K=2) in VALU producing exact A fragments; layer2 via WMMA with
// B fragments read from TDM-staged LDS (row stride 65 dwords).
// ---------------------------------------------------------------------------
struct ConvK {
    const float*          W1;    // 2 x 128 f32
    const float*          b1;    // 128 f32
    const unsigned short* W2t;   // out x 128 f16 (transposed, padded K)
    const float*          b2;    // out f32
    float*                agg;   // n_dst x out
};

__global__ void edge_conv16_kernel(const float2* __restrict__ off,
                                   const int* __restrict__ src,
                                   const int* __restrict__ dst,
                                   const float* __restrict__ ex,
                                   const float* __restrict__ denom,
                                   const float* __restrict__ feat,
                                   ConvK c0, ConvK c1, ConvK c2, int E) {
    __shared__ float    s_w[8][16];
    __shared__ int      s_src[8][16];
    __shared__ int      s_dst[8][16];
    __shared__ unsigned s_w2t[3][16 * 65];   // 16 rows x 65 dwords (padded)

    const int lane = threadIdx.x & 31;
    const int wave = threadIdx.x >> 5;
    const int lr   = lane & 15;
    const int half = lane >> 4;
    const int base = blockIdx.x * 128 + wave * 16;
    const int e    = base + lr;
    const bool ok  = (e < E);

    ConvK cs[3] = {c0, c1, c2};
    if (wave == 0) {
        tdm_load_w2_f16((unsigned)(size_t)&s_w2t[0][0], cs[0].W2t, 16u);
        tdm_load_w2_f16((unsigned)(size_t)&s_w2t[1][0], cs[1].W2t, 16u);
        tdm_load_w2_f16((unsigned)(size_t)&s_w2t[2][0], cs[2].W2t, 16u);
        __builtin_amdgcn_s_wait_tensorcnt(0);
    }

    float ox = 0.f, oy = 0.f;
    if (ok) { float2 o = off[e]; ox = o.x; oy = o.y; }
    if (half == 0) {
        float w = 0.f; int s = 0, d = 0;
        if (ok) {
            d = dst[e]; s = src[e];
            w = ex[e] / fmaxf(denom[d], 1e-20f);
        }
        s_w[wave][lr] = w; s_src[wave][lr] = s; s_dst[wave][lr] = d;
    }
    __syncthreads();

    for (int j = 0; j < 3; ++j) {
        const ConvK ck = cs[j];
        v16h afr[4];
#pragma unroll
        for (int kt = 0; kt < 4; ++kt) {
            const int ks = kt * 32 + half * 16;
#pragma unroll
            for (int i = 0; i < 16; ++i) {
                const int k = ks + i;   // {ks..ks+15}
                float h = elu_(ox * ck.W1[k] + oy * ck.W1[128 + k] + ck.b1[k]);
                afr[kt][i] = (_Float16)h;
            }
        }
        v8f acc = {};
#pragma unroll
        for (int kt = 0; kt < 4; ++kt) {
            v16h b;
            const int bd = lr * 65 + kt * 16 + half * 8;  // row lr, dword kb/2
#pragma unroll
            for (int i = 0; i < 8; ++i) {
                HalfPack p; p.u = s_w2t[j][bd + i];
                b[2 * i]     = p.h[0];
                b[2 * i + 1] = p.h[1];
            }
            acc = __builtin_amdgcn_wmma_f32_16x16x32_f16(false, afr[kt], false, b,
                                                         (short)0, acc, false, false);
        }
        const float bb = ck.b2[lr];
#pragma unroll
        for (int v = 0; v < 8; ++v) {
            const int r  = v + half * 8;
            const int ee = base + r;
            if (ee < E) {
                float kw = acc[v] + bb;
                float w  = s_w[wave][r];
                int   sE = s_src[wave][r];
                int   dE = s_dst[wave][r];
                atomicAdd(&ck.agg[dE * INF_ + lr], feat[sE * INF_ + lr] * (w * kw));
            }
        }
    }
}

// ---------------------------------------------------------------------------
// Fused edge conv level 1 (in_f = 128): per-conv feat (elu on load).
// ---------------------------------------------------------------------------
struct ConvK2 {
    const float*          W1;
    const float*          b1;
    const unsigned short* W2t;   // 128 x 128 f16 transposed
    const float*          b2;
    const float*          feat;  // N1 x 128 (pre-elu)
    float*                agg;   // N2 x 128
};

__global__ void edge_conv128_kernel(const float2* __restrict__ off,
                                    const int* __restrict__ src,
                                    const int* __restrict__ dst,
                                    const float* __restrict__ ex,
                                    const float* __restrict__ denom,
                                    ConvK2 c0, ConvK2 c1, ConvK2 c2, int E) {
    __shared__ float    s_w[8][16];
    __shared__ int      s_src[8][16];
    __shared__ int      s_dst[8][16];
    __shared__ unsigned s_w2t[3][128 * 65];  // 128 rows x 65 dwords (padded)

    const int lane = threadIdx.x & 31;
    const int wave = threadIdx.x >> 5;
    const int lr   = lane & 15;
    const int half = lane >> 4;
    const int base = blockIdx.x * 128 + wave * 16;
    const int e    = base + lr;
    const bool ok  = (e < E);

    ConvK2 cs[3] = {c0, c1, c2};
    if (wave == 0) {
        tdm_load_w2_f16((unsigned)(size_t)&s_w2t[0][0], cs[0].W2t, 128u);
        tdm_load_w2_f16((unsigned)(size_t)&s_w2t[1][0], cs[1].W2t, 128u);
        tdm_load_w2_f16((unsigned)(size_t)&s_w2t[2][0], cs[2].W2t, 128u);
        __builtin_amdgcn_s_wait_tensorcnt(0);
    }

    float ox = 0.f, oy = 0.f;
    if (ok) { float2 o = off[e]; ox = o.x; oy = o.y; }
    if (half == 0) {
        float w = 0.f; int s = 0, d = 0;
        if (ok) {
            d = dst[e]; s = src[e];
            w = ex[e] / fmaxf(denom[d], 1e-20f);
        }
        s_w[wave][lr] = w; s_src[wave][lr] = s; s_dst[wave][lr] = d;
    }
    __syncthreads();

    for (int j = 0; j < 3; ++j) {
        const ConvK2 ck = cs[j];
        v16h afr[4];
#pragma unroll
        for (int kt = 0; kt < 4; ++kt) {
            const int ks = kt * 32 + half * 16;
#pragma unroll
            for (int i = 0; i < 16; ++i) {
                const int k = ks + i;
                float h = elu_(ox * ck.W1[k] + oy * ck.W1[128 + k] + ck.b1[k]);
                afr[kt][i] = (_Float16)h;
            }
        }
        for (int nt = 0; nt < 8; ++nt) {
            v8f acc = {};
#pragma unroll
            for (int kt = 0; kt < 4; ++kt) {
                v16h b;
                const int bd = (nt * 16 + lr) * 65 + kt * 16 + half * 8;
#pragma unroll
                for (int i = 0; i < 8; ++i) {
                    HalfPack p; p.u = s_w2t[j][bd + i];
                    b[2 * i]     = p.h[0];
                    b[2 * i + 1] = p.h[1];
                }
                acc = __builtin_amdgcn_wmma_f32_16x16x32_f16(false, afr[kt], false, b,
                                                             (short)0, acc, false, false);
            }
            const int   col = nt * 16 + lr;
            const float bb  = ck.b2[col];
#pragma unroll
            for (int v = 0; v < 8; ++v) {
                const int r  = v + half * 8;
                const int ee = base + r;
                if (ee < E) {
                    float kw = acc[v] + bb;
                    float w  = s_w[wave][r];
                    int   sE = s_src[wave][r];
                    int   dE = s_dst[wave][r];
                    atomicAdd(&ck.agg[dE * H_ + col],
                              elu_(ck.feat[sE * H_ + col]) * (w * kw));
                }
            }
        }
    }
}

// ---------------------------------------------------------------------------
// Generic WMMA GEMM: C = actOut([A0|A1] @ W + bias (+ skip)); W pre-converted
// to f16 transposed (N x Kpad) -> B fragment is one 32B contiguous load.
// ---------------------------------------------------------------------------
__global__ void gemm_wmma_kernel(const float* __restrict__ A0, int K0,
                                 const float* __restrict__ A1, int K1,
                                 const unsigned short* __restrict__ Wt, int Kpad,
                                 const float* __restrict__ bias,
                                 const float* __restrict__ skip, int ldSkip,
                                 float* __restrict__ C, int ldC, int cOff,
                                 int M, int N, int actA, int actOut) {
    const int lane = threadIdx.x & 31;
    const int wave = threadIdx.x >> 5;
    const int mt   = blockIdx.y * 8 + wave;
    const int nt   = blockIdx.x;
    const int row0 = mt * 16;
    if (row0 >= M) return;

    const int K    = K0 + K1;
    const int half = lane >> 4;
    const int lr   = lane & 15;
    const int col  = nt * 16 + lr;
    const int arow = row0 + lr;

    v8f acc = {};
    const int nkt = Kpad >> 5;
    for (int kt = 0; kt < nkt; ++kt) {
        v16h a;
        const int ksA = kt * 32 + half * 16;
#pragma unroll
        for (int i = 0; i < 16; ++i) {
            const int k = ksA + i;
            float v = 0.f;
            if (arow < M && k < K) {
                v = (k < K0) ? A0[arow * K0 + k] : A1[arow * K1 + (k - K0)];
                if (actA) v = elu_(v);
            }
            a[i] = (_Float16)v;
        }
        v16h b = {};
        if (col < N)
            b = *(const v16h*)(Wt + (size_t)col * Kpad + kt * 32 + half * 16);
        acc = __builtin_amdgcn_wmma_f32_16x16x32_f16(false, a, false, b,
                                                     (short)0, acc, false, false);
    }
#pragma unroll
    for (int v = 0; v < 8; ++v) {
        const int row = row0 + v + half * 8;
        if (row < M && col < N) {
            float x = acc[v] + (bias ? bias[col] : 0.f);
            if (skip) x += skip[row * ldSkip + col];
            if (actOut) x = elu_(x);
            C[row * ldC + cOff + col] = x;
        }
    }
}

// ---------------------------------------------------------------------------
// Host
// ---------------------------------------------------------------------------
extern "C" void kernel_launch(void* const* d_in, const int* in_sizes, int n_in,
                              void* d_out, int out_size, void* d_ws, size_t ws_size,
                              hipStream_t stream) {
    (void)in_sizes; (void)n_in; (void)out_size; (void)ws_size;

    const float*  feat = (const float*)d_in[0];
    const float2* off0 = (const float2*)d_in[1];
    const float2* off1 = (const float2*)d_in[2];
    const int*    src0 = (const int*)d_in[3];
    const int*    dst0 = (const int*)d_in[4];
    const int*    src1 = (const int*)d_in[5];
    const int*    dst1 = (const int*)d_in[6];
    // d_in[7], d_in[8]: n_dst0 / n_dst1 (compile-time constants here)

    // params: pytree leaf order (sorted dict keys; lin = {b, w})
    const int P = 9;
    auto F = [&](int i) { return (const float*)d_in[i]; };
    auto CV = [&](int i) { return P + i * 16; };  // conv_i base, i=0..5
    const int O1 = P + 96, O2 = P + 104, O3 = P + 114, SK = P + 120;

    // ---- workspace: f32 region ----
    float* ws = (float*)d_ws;
    size_t o = 0;
    unsigned* smax0 = (unsigned*)(ws + o); o += N1_;
    float* denom0 = ws + o; o += N1_;
    float* agg0[3]; for (int j = 0; j < 3; ++j) { agg0[j] = ws + o; o += (size_t)N1_ * INF_; }
    unsigned* smax1 = (unsigned*)(ws + o); o += N2_;
    float* denom1 = ws + o; o += N2_;
    float* agg1[3]; for (int j = 0; j < 3; ++j) { agg1[j] = ws + o; o += (size_t)N2_ * H_; }
    float* ex0 = ws + o; o += E0_;
    float* ex1 = ws + o; o += E1_;
    float* hbuf[3]; for (int j = 0; j < 3; ++j) { hbuf[j] = ws + o; o += (size_t)N1_ * H_; }
    float* h2buf[3]; for (int j = 0; j < 3; ++j) { h2buf[j] = ws + o; o += (size_t)N2_ * H_; }
    float* t_buf = ws + o; o += (size_t)N1_ * H_;
    float* hsb   = ws + o; o += (size_t)N1_ * H_;
    float* t2    = ws + o; o += (size_t)N1_ * H_;
    float* t3    = ws + o; o += (size_t)N1_ * H_;
    float* skipb = ws + o; o += (size_t)N1_ * H_;
    o = (o + 15) & ~(size_t)15;                    // 64B-align f16 region
    unsigned short* f16ws = (unsigned short*)(ws + o);
    size_t f16o = 0;

    // zero the atomic accumulators (two contiguous regions)
    hipMemsetAsync(smax0, 0, (size_t)(2 * N1_ + 3 * N1_ * INF_) * 4, stream);
    hipMemsetAsync(smax1, 0, (size_t)(2 * N2_ + 3 * (size_t)N2_ * H_) * 4, stream);

    // ---- pre-convert every weight matrix to transposed padded f16 ----
    auto cvt = [&](const float* W, int K, int N) -> const unsigned short* {
        int Kpad = (K + 31) & ~31;
        unsigned short* dstp = f16ws + f16o;
        f16o += (size_t)N * Kpad;
        int total = N * Kpad;
        wconv_kernel<<<(total + 255) / 256, 256, 0, stream>>>(W, dstp, K, N, Kpad);
        return dstp;
    };

    const unsigned short *kW2t[6], *sW0t[6], *sW1t[6], *mWt[6][4], *skWt[6];
    for (int j = 0; j < 6; ++j) {
        int c = CV(j), inf = (j < 3) ? INF_ : H_;
        kW2t[j]   = cvt(F(c + 3), H_, inf);           // kernel.w1: 128 x inf
        sW0t[j]   = cvt(F(c + 13), inf, H_);          // mlp_self.w0
        sW1t[j]   = cvt(F(c + 15), H_, H_);           // mlp_self.w1
        mWt[j][0] = cvt(F(c + 5), inf + H_, H_);      // mlp.w0
        mWt[j][1] = cvt(F(c + 7), H_, H_);
        mWt[j][2] = cvt(F(c + 9), H_, H_);
        mWt[j][3] = cvt(F(c + 11), H_, H_);
        skWt[j]   = cvt(F(SK + 2 * j + 1), INF_, H_); // skip.w
    }
    const unsigned short* o1W[4] = { cvt(F(O1 + 1), 2 * H_, H_), cvt(F(O1 + 3), H_, H_),
                                     cvt(F(O1 + 5), H_, H_),     cvt(F(O1 + 7), H_, 2) };
    const unsigned short* o2W[5] = { cvt(F(O2 + 1), 2 * H_, H_), cvt(F(O2 + 3), H_, H_),
                                     cvt(F(O2 + 5), H_, H_),     cvt(F(O2 + 7), H_, H_),
                                     cvt(F(O2 + 9), H_, 1) };
    const unsigned short* o3W[3] = { cvt(F(O3 + 1), 2 * H_, H_), cvt(F(O3 + 3), H_, H_),
                                     cvt(F(O3 + 5), H_, 1) };

    auto gemm = [&](const float* A0, int K0, const float* A1, int K1,
                    const unsigned short* Wt, const float* b,
                    const float* skip, int ldSkip,
                    float* C, int ldC, int cOff, int M, int N, int actA, int actOut) {
        int Kpad = (K0 + K1 + 31) & ~31;
        dim3 g((N + 15) / 16, (M + 127) / 128);
        gemm_wmma_kernel<<<g, 256, 0, stream>>>(A0, K0, A1, K1, Wt, Kpad, b, skip, ldSkip,
                                                C, ldC, cOff, M, N, actA, actOut);
    };

    // ===================== level 0 softmax + edge conv =====================
    score_max_kernel<<<(E0_ + 255) / 256, 256, 0, stream>>>(off0, dst0, smax0, E0_);
    score_exp_kernel<<<(E0_ + 255) / 256, 256, 0, stream>>>(off0, dst0, smax0, denom0, ex0, E0_);
    {
        ConvK ck[3];
        for (int j = 0; j < 3; ++j) {
            int c = CV(j);
            ck[j] = ConvK{ F(c + 1), F(c + 0), kW2t[j], F(c + 2), agg0[j] };
        }
        edge_conv16_kernel<<<(E0_ + 127) / 128, 256, 0, stream>>>(
            off0, src0, dst0, ex0, denom0, feat, ck[0], ck[1], ck[2], E0_);
    }

    // ===================== level 0 node MLPs =====================
    for (int j = 0; j < 3; ++j) {
        int c = CV(j);
        gemm(feat, INF_, nullptr, 0, sW0t[j], F(c + 12), nullptr, 0, t_buf, H_, 0, N1_, H_, 0, 1);
        gemm(t_buf, H_, nullptr, 0, sW1t[j], F(c + 14), nullptr, 0, hsb, H_, 0, N1_, H_, 0, 0);
        gemm(feat, INF_, nullptr, 0, skWt[j], F(SK + 2 * j), nullptr, 0, skipb, H_, 0, N1_, H_, 0, 0);
        gemm(agg0[j], INF_, hsb, H_, mWt[j][0], F(c + 4), nullptr, 0, t2, H_, 0, N1_, H_, 0, 1);
        gemm(t2, H_, nullptr, 0, mWt[j][1], F(c + 6), nullptr, 0, t3, H_, 0, N1_, H_, 0, 1);
        gemm(t3, H_, nullptr, 0, mWt[j][2], F(c + 8), nullptr, 0, t2, H_, 0, N1_, H_, 0, 1);
        gemm(t2, H_, nullptr, 0, mWt[j][3], F(c + 10), skipb, H_, hbuf[j], H_, 0, N1_, H_, 0, 0);
    }

    // ===================== level 1 softmax + edge conv =====================
    score_max_kernel<<<(E1_ + 255) / 256, 256, 0, stream>>>(off1, dst1, smax1, E1_);
    score_exp_kernel<<<(E1_ + 255) / 256, 256, 0, stream>>>(off1, dst1, smax1, denom1, ex1, E1_);
    {
        ConvK2 ck[3];
        for (int j = 0; j < 3; ++j) {
            int c = CV(3 + j);
            ck[j] = ConvK2{ F(c + 1), F(c + 0), kW2t[3 + j], F(c + 2), hbuf[j], agg1[j] };
        }
        edge_conv128_kernel<<<(E1_ + 127) / 128, 256, 0, stream>>>(
            off1, src1, dst1, ex1, denom1, ck[0], ck[1], ck[2], E1_);
    }

    // ===================== level 1 node MLPs (+ h2 skips) =====================
    for (int j = 0; j < 3; ++j) {
        int c = CV(3 + j), jj = 3 + j;
        gemm(hbuf[j], H_, nullptr, 0, sW0t[jj], F(c + 12), nullptr, 0, t_buf, H_, 0, N2_, H_, 1, 1);
        gemm(t_buf, H_, nullptr, 0, sW1t[jj], F(c + 14), nullptr, 0, hsb, H_, 0, N2_, H_, 0, 0);
        gemm(agg1[j], H_, hsb, H_, mWt[jj][0], F(c + 4), nullptr, 0, t2, H_, 0, N2_, H_, 0, 1);
        gemm(t2, H_, nullptr, 0, mWt[jj][1], F(c + 6), nullptr, 0, t3, H_, 0, N2_, H_, 0, 1);
        gemm(t3, H_, nullptr, 0, mWt[jj][2], F(c + 8), nullptr, 0, t2, H_, 0, N2_, H_, 0, 1);
        gemm(t2, H_, nullptr, 0, mWt[jj][3], F(c + 10), hbuf[j], H_, hbuf[j], H_, 0, N2_, H_, 0, 0);
        gemm(feat, INF_, nullptr, 0, skWt[jj], F(SK + 2 * jj), nullptr, 0, h2buf[j], H_, 0, N2_, H_, 0, 0);
    }

    float* out = (float*)d_out;  // N2 x 4 : [out1(2) | out2(1) | out3(1)]

    // head 1: 256 ->128->128->128-> 2
    gemm(hbuf[0], H_, h2buf[0], H_, o1W[0], F(O1 + 0), nullptr, 0, t_buf, H_, 0, N2_, H_, 1, 1);
    gemm(t_buf, H_, nullptr, 0, o1W[1], F(O1 + 2), nullptr, 0, t2, H_, 0, N2_, H_, 0, 1);
    gemm(t2, H_, nullptr, 0, o1W[2], F(O1 + 4), nullptr, 0, t3, H_, 0, N2_, H_, 0, 1);
    gemm(t3, H_, nullptr, 0, o1W[3], F(O1 + 6), nullptr, 0, out, 4, 0, N2_, 2, 0, 0);

    // head 2: 256 ->128->128->128->128-> 1
    gemm(hbuf[1], H_, h2buf[1], H_, o2W[0], F(O2 + 0), nullptr, 0, t_buf, H_, 0, N2_, H_, 1, 1);
    gemm(t_buf, H_, nullptr, 0, o2W[1], F(O2 + 2), nullptr, 0, t2, H_, 0, N2_, H_, 0, 1);
    gemm(t2, H_, nullptr, 0, o2W[2], F(O2 + 4), nullptr, 0, t3, H_, 0, N2_, H_, 0, 1);
    gemm(t3, H_, nullptr, 0, o2W[3], F(O2 + 6), nullptr, 0, t_buf, H_, 0, N2_, H_, 0, 1);
    gemm(t_buf, H_, nullptr, 0, o2W[4], F(O2 + 8), nullptr, 0, out, 4, 2, N2_, 1, 0, 0);

    // head 3: 256 ->128->128-> 1
    gemm(hbuf[2], H_, h2buf[2], H_, o3W[0], F(O3 + 0), nullptr, 0, t_buf, H_, 0, N2_, H_, 1, 1);
    gemm(t_buf, H_, nullptr, 0, o3W[1], F(O3 + 2), nullptr, 0, t2, H_, 0, N2_, H_, 0, 1);
    gemm(t2, H_, nullptr, 0, o3W[2], F(O3 + 4), nullptr, 0, out, 4, 3, N2_, 1, 0, 0);
}
